// mplstm_48584670053219
// MI455X (gfx1250) — matrix operand order repaired and checked by reading, not compiled, not run on hardware
//
#include <hip/hip_runtime.h>
#include <hip/hip_bf16.h>

typedef __attribute__((ext_vector_type(16))) _Float16 v16h;
typedef __attribute__((ext_vector_type(8)))  float    v8f;

#define D_DIM   512
#define B_DIM   2
#define LQ      512
#define LK      712
#define NCACHE  200
#define KSZ     5
#define TOUT    508
#define CPAD    224   // padded cache dim (multiple of 32 for WMMA K-loop)

union F16Frag { uint4 u4[2]; v16h h; };

__device__ inline v16h zero16() {
  F16Frag f;
  f.u4[0] = make_uint4(0u, 0u, 0u, 0u);
  f.u4[1] = make_uint4(0u, 0u, 0u, 0u);
  return f.h;
}

// A-matrix fragment, 16x32 f16 tile at (row, k0). ISA 7.12.2 layout:
// lane<16: K = k0..k0+7 (elems 0..7), k0+16..k0+23 (elems 8..15)
// lane>=16: same +8.
__device__ inline v16h ld_frag_a(const _Float16* X, int ld, int row, int k0, int lane) {
  F16Frag f;
  const _Float16* p = X + (size_t)row * ld + k0 + ((lane >> 4) * 8);
  f.u4[0] = *reinterpret_cast<const uint4*>(p);
  f.u4[1] = *reinterpret_cast<const uint4*>(p + 16);
  return f.h;
}

// B-matrix fragment for B = W^T (32x16): lane n holds row n of W,
// K = k0..k0+7 (elems 0..7), k0+16..k0+23 (elems 8..15); lane>=16: same +8.
__device__ inline v16h ld_frag_b(const _Float16* W, int ld, int nrow, int k0, int lane) {
  F16Frag f;
  const _Float16* p = W + (size_t)nrow * ld + k0 + ((lane >> 4) * 8);
  f.u4[0] = *reinterpret_cast<const uint4*>(p);
  f.u4[1] = *reinterpret_cast<const uint4*>(p + 16);
  return f.h;
}

#define WMMA_F16(A, B, C) \
  __builtin_amdgcn_wmma_f32_16x16x32_f16(false, (A), false, (B), (short)0, (C), false, false)

// ---------------- LayerNorm -> f16 ----------------
__global__ void ln_f16_kernel(const float* __restrict__ x, const float* __restrict__ g,
                              const float* __restrict__ b, _Float16* __restrict__ y) {
  __shared__ float red[256];
  const int row = blockIdx.x;
  const int tid = threadIdx.x;
  const float* xr = x + (size_t)row * D_DIM;
  float x0 = xr[tid], x1 = xr[tid + 256];
  red[tid] = x0 + x1;
  __syncthreads();
  for (int s = 128; s > 0; s >>= 1) { if (tid < s) red[tid] += red[tid + s]; __syncthreads(); }
  float mu = red[0] * (1.0f / (float)D_DIM);
  __syncthreads();
  float d0 = x0 - mu, d1 = x1 - mu;
  red[tid] = d0 * d0 + d1 * d1;
  __syncthreads();
  for (int s = 128; s > 0; s >>= 1) { if (tid < s) red[tid] += red[tid + s]; __syncthreads(); }
  float rstd = rsqrtf(red[0] * (1.0f / (float)D_DIM) + 1e-5f);
  _Float16* yr = y + (size_t)row * D_DIM;
  yr[tid]       = (_Float16)(d0 * rstd * g[tid]       + b[tid]);
  yr[tid + 256] = (_Float16)(d1 * rstd * g[tid + 256] + b[tid + 256]);
}

// ---------------- weight converts ----------------
__global__ void f32_to_f16_kernel(const float* __restrict__ s, _Float16* __restrict__ d, int n) {
  for (int i = blockIdx.x * blockDim.x + threadIdx.x; i < n; i += gridDim.x * blockDim.x)
    d[i] = (_Float16)s[i];
}

// conv_w (200, 512, 5) -> cw16[kk][o][d], o padded to CPAD (zeros beyond 200)
__global__ void conv_repack_kernel(const float* __restrict__ s, _Float16* __restrict__ d) {
  const int n = KSZ * CPAD * D_DIM;
  for (int i = blockIdx.x * blockDim.x + threadIdx.x; i < n; i += gridDim.x * blockDim.x) {
    int kk = i / (CPAD * D_DIM);
    int rem = i - kk * (CPAD * D_DIM);
    int o = rem / D_DIM;
    int dd = rem - o * D_DIM;
    float v = (o < NCACHE) ? s[((size_t)o * D_DIM + dd) * KSZ + kk] : 0.0f;
    d[i] = (_Float16)v;
  }
}

// fnn_w (200,200) -> fn16 (CPAD x CPAD), zero padded
__global__ void fnn_pad_kernel(const float* __restrict__ s, _Float16* __restrict__ d) {
  const int n = CPAD * CPAD;
  for (int i = blockIdx.x * blockDim.x + threadIdx.x; i < n; i += gridDim.x * blockDim.x) {
    int o = i / CPAD, j = i - o * CPAD;
    float v = (o < NCACHE && j < NCACHE) ? s[o * NCACHE + j] : 0.0f;
    d[i] = (_Float16)v;
  }
}

// ---------------- fused k/v projection: k = relu(Kn@wk^T+bk) * (Vn@wv^T+bv) ----------------
// One wave computes a 16x64 strip: one A fragment feeds 4 B tiles (x2 gemms).
__global__ void gemm_kv_kernel(const _Float16* __restrict__ kn, const _Float16* __restrict__ vn,
                               const _Float16* __restrict__ wk, const _Float16* __restrict__ wv,
                               const float* __restrict__ bk_b, const float* __restrict__ bv_b,
                               float* __restrict__ kout) {
  const int lane = threadIdx.x;
  const int m0 = blockIdx.x * 16;
  const int n0 = blockIdx.y * 64;
  const int arow = m0 + (lane & 15);
  const int bbase = n0 + (lane & 15);
  v8f acck[4] = {}, accv[4] = {};
  for (int k0 = 0; k0 < D_DIM; k0 += 32) {
    v16h ak = ld_frag_a(kn, D_DIM, arow, k0, lane);
    v16h av = ld_frag_a(vn, D_DIM, arow, k0, lane);
    #pragma unroll
    for (int n = 0; n < 4; ++n) {
      v16h bkf = ld_frag_b(wk, D_DIM, bbase + n * 16, k0, lane);
      acck[n] = WMMA_F16(ak, bkf, acck[n]);
      v16h bvf = ld_frag_b(wv, D_DIM, bbase + n * 16, k0, lane);
      accv[n] = WMMA_F16(av, bvf, accv[n]);
    }
  }
  const int hs = lane >> 4;
  #pragma unroll
  for (int n = 0; n < 4; ++n) {
    int col = n0 + n * 16 + (lane & 15);
    float bkc = bk_b[col], bvc = bv_b[col];
    for (int r = 0; r < 8; ++r) {
      int row = m0 + r + hs * 8;
      kout[(size_t)row * D_DIM + col] =
          fmaxf(acck[n][r] + bkc, 0.0f) * (accv[n][r] + bvc);
    }
  }
}

// ---------------- q projection -> f16 (16x64 strip per wave) ----------------
__global__ void gemm_q_kernel(const _Float16* __restrict__ qn, const _Float16* __restrict__ wq,
                              const float* __restrict__ bq, _Float16* __restrict__ qg) {
  const int lane = threadIdx.x;
  const int m0 = blockIdx.x * 16;
  const int n0 = blockIdx.y * 64;
  const int arow = m0 + (lane & 15);
  const int bbase = n0 + (lane & 15);
  v8f acc[4] = {};
  for (int k0 = 0; k0 < D_DIM; k0 += 32) {
    v16h a = ld_frag_a(qn, D_DIM, arow, k0, lane);
    #pragma unroll
    for (int n = 0; n < 4; ++n) {
      v16h b = ld_frag_b(wq, D_DIM, bbase + n * 16, k0, lane);
      acc[n] = WMMA_F16(a, b, acc[n]);
    }
  }
  const int hs = lane >> 4;
  #pragma unroll
  for (int n = 0; n < 4; ++n) {
    int col = n0 + n * 16 + (lane & 15);
    float bc = bq[col];
    for (int r = 0; r < 8; ++r) {
      int row = m0 + r + hs * 8;
      qg[(size_t)row * D_DIM + col] = (_Float16)(acc[n][r] + bc);
    }
  }
}

// ---------------- conv as 5 shifted GEMM passes: c[b,t,o] (16x32 strip) ----------------
__global__ void conv_gemm_kernel(const _Float16* __restrict__ qg, const _Float16* __restrict__ cw,
                                 const float* __restrict__ cb, _Float16* __restrict__ cp) {
  const int lane = threadIdx.x;
  const int b = blockIdx.x >> 5;             // 32 t-tiles per batch
  const int t0 = (blockIdx.x & 31) * 16;
  const int n0 = blockIdx.y * 32;            // o-strip (7 strips = CPAD)
  const _Float16* q = qg + (size_t)b * LQ * D_DIM;
  const int am = lane & 15;
  const int bbase = n0 + am;
  v8f acc[2] = {};
  for (int kk = 0; kk < KSZ; ++kk) {
    const int arow = t0 + am + kk;
    const bool avalid = (arow < LQ);
    const _Float16* wkk = cw + (size_t)kk * CPAD * D_DIM;
    for (int k0 = 0; k0 < D_DIM; k0 += 32) {
      v16h a = avalid ? ld_frag_a(q, D_DIM, arow, k0, lane) : zero16();
      #pragma unroll
      for (int n = 0; n < 2; ++n) {
        v16h w = ld_frag_b(wkk, D_DIM, bbase + n * 16, k0, lane);
        acc[n] = WMMA_F16(a, w, acc[n]);
      }
    }
  }
  const int hs = lane >> 4;
  #pragma unroll
  for (int n = 0; n < 2; ++n) {
    int col = n0 + n * 16 + (lane & 15);
    float bias = (col < NCACHE) ? cb[col] : 0.0f;
    for (int r = 0; r < 8; ++r) {
      int t = t0 + r + hs * 8;
      float v = fmaxf(acc[n][r] + bias, 0.0f);
      cp[((size_t)b * LQ + t) * CPAD + col] = (_Float16)v;
    }
  }
}

// ---------------- w = sigmoid(c @ fnn^T + fb) ----------------
__global__ void fnn_gemm_kernel(const _Float16* __restrict__ cp, const _Float16* __restrict__ fn,
                                const float* __restrict__ fb, float* __restrict__ wout) {
  const int lane = threadIdx.x;
  const int b = blockIdx.x >> 5;
  const int t0 = (blockIdx.x & 31) * 16;
  const int n0 = blockIdx.y * 16;            // 13 tiles cover 208 >= 200
  const _Float16* c = cp + (size_t)b * LQ * CPAD;
  const int arow = t0 + (lane & 15);
  const int brow = n0 + (lane & 15);
  v8f acc = {};
  for (int k0 = 0; k0 < CPAD; k0 += 32) {
    v16h a = ld_frag_a(c, CPAD, arow, k0, lane);
    v16h w = ld_frag_b(fn, CPAD, brow, k0, lane);
    acc = WMMA_F16(a, w, acc);
  }
  const int col = n0 + (lane & 15);
  if (col < NCACHE) {
    const float bias = fb[col];
    const int hs = lane >> 4;
    for (int r = 0; r < 8; ++r) {
      int t = t0 + r + hs * 8;
      float x = acc[r] + bias;
      wout[((size_t)b * LQ + t) * NCACHE + col] = 1.0f / (1.0f + __expf(-x));
    }
  }
}

// ---------------- banded einsum, t-tiled to share the k window ----------------
// out[b,t,d] = sum_j k[b,t+j,d]*w[b,t,j] / 200
// Block: 128 threads (one d each), 16 t-values per block; w tile staged in LDS
// (broadcast reads). Each k row in the 215-row window is loaded once per t-tile.
__global__ void einsum_tiled_kernel(const float* __restrict__ kf, const float* __restrict__ wf,
                                    float* __restrict__ out) {
  __shared__ float wsh[16][NCACHE];
  const int b = blockIdx.x >> 5;
  const int t0 = (blockIdx.x & 31) * 16;
  const int d = blockIdx.y * 128 + threadIdx.x;
  const int tid = threadIdx.x;
  for (int i = tid; i < 16 * NCACHE; i += 128) {
    int tt = i / NCACHE, j = i - tt * NCACHE;
    wsh[tt][j] = wf[((size_t)b * LQ + t0 + tt) * NCACHE + j];
  }
  __syncthreads();
  float acc[16];
  #pragma unroll
  for (int i = 0; i < 16; ++i) acc[i] = 0.0f;
  const float* kb = kf + (size_t)b * LK * D_DIM;
  for (int s = t0; s < t0 + 16 + NCACHE - 1; ++s) {   // 215-row window
    float kv = kb[(size_t)s * D_DIM + d];
    int ttlo = s - t0 - (NCACHE - 1); ttlo = ttlo < 0 ? 0 : ttlo;
    int tthi = s - t0;                tthi = tthi > 15 ? 15 : tthi;
    for (int tt = ttlo; tt <= tthi; ++tt)
      acc[tt] += kv * wsh[tt][s - t0 - tt];
  }
  #pragma unroll
  for (int tt = 0; tt < 16; ++tt) {
    int t = t0 + tt;
    if (t < TOUT)
      out[((size_t)b * TOUT + t) * D_DIM + d] = acc[tt] * (1.0f / (float)NCACHE);
  }
}

extern "C" void kernel_launch(void* const* d_in, const int* in_sizes, int n_in,
                              void* d_out, int out_size, void* d_ws, size_t ws_size,
                              hipStream_t stream) {
  const float* query  = (const float*)d_in[0];
  const float* key    = (const float*)d_in[1];
  const float* value  = (const float*)d_in[2];
  const float* lnq_g  = (const float*)d_in[3];
  const float* lnq_b  = (const float*)d_in[4];
  const float* lnk_g  = (const float*)d_in[5];
  const float* lnk_b  = (const float*)d_in[6];
  const float* lnv_g  = (const float*)d_in[7];
  const float* lnv_b  = (const float*)d_in[8];
  const float* wq_w   = (const float*)d_in[9];
  const float* wq_b   = (const float*)d_in[10];
  const float* wk_w   = (const float*)d_in[11];
  const float* wk_b   = (const float*)d_in[12];
  const float* wv_w   = (const float*)d_in[13];
  const float* wv_b   = (const float*)d_in[14];
  const float* conv_w = (const float*)d_in[15];
  const float* conv_b = (const float*)d_in[16];
  const float* fnn_w  = (const float*)d_in[17];
  const float* fnn_b  = (const float*)d_in[18];
  float* out = (float*)d_out;

  char* ws = (char*)d_ws;
  size_t off = 0;
  auto carve = [&](size_t bytes) -> void* {
    void* p = ws + off;
    off += (bytes + 255) & ~(size_t)255;
    return p;
  };
  const int MKV = B_DIM * LK;   // 1424, divisible by 16
  const int MQ  = B_DIM * LQ;   // 1024
  _Float16* kn16 = (_Float16*)carve((size_t)MKV * D_DIM * 2);
  _Float16* vn16 = (_Float16*)carve((size_t)MKV * D_DIM * 2);
  _Float16* qn16 = (_Float16*)carve((size_t)MQ * D_DIM * 2);
  _Float16* wk16 = (_Float16*)carve((size_t)D_DIM * D_DIM * 2);
  _Float16* wv16 = (_Float16*)carve((size_t)D_DIM * D_DIM * 2);
  _Float16* wq16 = (_Float16*)carve((size_t)D_DIM * D_DIM * 2);
  _Float16* cw16 = (_Float16*)carve((size_t)KSZ * CPAD * D_DIM * 2);
  _Float16* fn16 = (_Float16*)carve((size_t)CPAD * CPAD * 2);
  float*    kf32 = (float*)carve((size_t)MKV * D_DIM * 4);
  _Float16* qg16 = (_Float16*)carve((size_t)MQ * D_DIM * 2);
  _Float16* cp16 = (_Float16*)carve((size_t)B_DIM * LQ * CPAD * 2);
  float*    wf32 = (float*)carve((size_t)B_DIM * LQ * NCACHE * 4);

  // 1) layernorms -> f16
  ln_f16_kernel<<<MKV, 256, 0, stream>>>(key, lnk_g, lnk_b, kn16);
  ln_f16_kernel<<<MKV, 256, 0, stream>>>(value, lnv_g, lnv_b, vn16);
  ln_f16_kernel<<<MQ, 256, 0, stream>>>(query, lnq_g, lnq_b, qn16);

  // 2) weight converts / repacks
  f32_to_f16_kernel<<<256, 256, 0, stream>>>(wk_w, wk16, D_DIM * D_DIM);
  f32_to_f16_kernel<<<256, 256, 0, stream>>>(wv_w, wv16, D_DIM * D_DIM);
  f32_to_f16_kernel<<<256, 256, 0, stream>>>(wq_w, wq16, D_DIM * D_DIM);
  conv_repack_kernel<<<256, 256, 0, stream>>>(conv_w, cw16);
  fnn_pad_kernel<<<64, 256, 0, stream>>>(fnn_w, fn16);

  // 3) fused k/v projection (WMMA, 16x64 strips), k in fp32
  gemm_kv_kernel<<<dim3(MKV / 16, D_DIM / 64), 32, 0, stream>>>(
      kn16, vn16, wk16, wv16, wk_b, wv_b, kf32);

  // 4) q projection (WMMA, 16x64 strips) -> f16
  gemm_q_kernel<<<dim3(MQ / 16, D_DIM / 64), 32, 0, stream>>>(qn16, wq16, wq_b, qg16);

  // 5) conv via 5 shifted WMMA GEMM passes (16x32 strips) -> c (f16, padded)
  conv_gemm_kernel<<<dim3(B_DIM * (LQ / 16), CPAD / 32), 32, 0, stream>>>(
      qg16, cw16, conv_b, cp16);

  // 6) w = sigmoid(c @ fnn^T + b) (WMMA)
  fnn_gemm_kernel<<<dim3(B_DIM * (LQ / 16), 13), 32, 0, stream>>>(cp16, fn16, fnn_b, wf32);

  // 7) banded weighted-sum epilogue, t-tiled (~15x less k traffic)
  einsum_tiled_kernel<<<dim3(B_DIM * (LQ / 16), D_DIM / 128), 128, 0, stream>>>(
      kf32, wf32, out);
}
